// BaselineGAT_14697378087238
// MI455X (gfx1250) — matrix operand order repaired and checked by reading, not compiled, not run on hardware
//
#include <hip/hip_runtime.h>

// GAT on MI455X: WMMA f16 GEMMs (f32 accum), async-DMA LDS staging,
// L2-resident atomic edge softmax/scatter.
// All per-call state lives in d_ws / d_out and is re-initialized every call.

typedef __attribute__((ext_vector_type(16))) _Float16 v16h;
typedef __attribute__((ext_vector_type(8)))  float    v8f;
typedef __attribute__((ext_vector_type(4)))  float    v4f;

#define NEG_SLOPE 0.2f

// Order-preserving float <-> uint32 encoding for atomicMax-based segment max.
__device__ __forceinline__ unsigned fenc(float f) {
  unsigned u = __float_as_uint(f);
  return (u & 0x80000000u) ? ~u : (u | 0x80000000u);
}
__device__ __forceinline__ float fdec(unsigned u) {
  return (u & 0x80000000u) ? __uint_as_float(u & 0x7FFFFFFFu) : __uint_as_float(~u);
}

// CDNA5 async DMA: copy 16 bytes global -> LDS, tracked by ASYNCcnt.
__device__ __forceinline__ void async_g2lds_b128(unsigned lds_byte_addr,
                                                 const void* gaddr) {
  asm volatile("global_load_async_to_lds_b128 %0, %1, off"
               :: "v"(lds_byte_addr), "v"(gaddr) : "memory");
}
__device__ __forceinline__ void wait_async0() {
  asm volatile("s_wait_asynccnt 0x0" ::: "memory");
}

__global__ void zero_f32(float* __restrict__ p, long n) {
  long i = (long)blockIdx.x * blockDim.x + threadIdx.x;
  if (i < n) p[i] = 0.0f;
}

// H[N,NC] = A[N,K] @ W[K,NC]. One wave per 16x16 output tile; NC/16 waves per
// block share a 16-row A strip (f32, DMA'd into LDS via async-to-LDS) and W^T
// (f16 in LDS). f32->f16 conversion of A folded into fragment assembly.
// WMMA A-frag (16-bit A 16x32): lanes<16 hold K {kk..kk+15},
// lanes>=16 hold K {kk+16..kk+31}; row M = lane&15.
// WMMA B-frag (16-bit B 32x16): col N = lane&15, K {kk..kk+15} (lanes<16) or
// {kk+16..kk+31} (lanes>=16), contiguous in W^T rows.
template <int K, int NC>
__global__ void __launch_bounds__(32 * (NC / 16)) gemm_wmma(
    const float* __restrict__ A, const float* __restrict__ W,
    float* __restrict__ H, int N) {
  extern __shared__ char ldsraw[];
  float*    sAf = (float*)ldsraw;                     // [16][K] f32 (DMA)
  _Float16* sWT = (_Float16*)(ldsraw + 16 * K * 4);   // [NC][K] f16 (W^T)

  const int tid  = threadIdx.x;
  const int nthr = 32 * (NC / 16);
  const int row0 = blockIdx.x * 16;

  // Async-DMA the 16xK f32 A strip into LDS, 16B per lane per issue.
  const unsigned ldsA = (unsigned)(size_t)sAf;
  const int nchunk = 16 * K / 4;  // number of 16-byte chunks
  for (int c = tid; c < nchunk; c += nthr) {
    int r = c / (K / 4);
    int k = (c - r * (K / 4)) * 4;
    if (row0 + r < N)
      async_g2lds_b128(ldsA + (unsigned)c * 16u,
                       A + (long)(row0 + r) * K + k);
  }
  // Stage W^T as f16: coalesced global read, strided LDS write.
  for (int i = tid; i < NC * K; i += nthr) {
    int k = i / NC, n = i - k * NC;
    sWT[n * K + k] = (_Float16)W[i];
  }
  wait_async0();
  __syncthreads();

  const int  wave = tid >> 5;
  const int  lane = tid & 31;
  const int  m    = lane & 15;
  const bool hi   = lane >= 16;
  const int  col  = wave * 16 + m;

  const float*    arow = sAf + m * K;
  const _Float16* bcol = sWT + col * K;
  const int ka = hi ? 16 : 0;
  const int kb = hi ? 16 : 0;

  v8f c = {};
#pragma unroll
  for (int kk = 0; kk < K; kk += 32) {
    v4f x0 = *(const v4f*)(arow + kk + ka);        // ds_load_b128
    v4f x1 = *(const v4f*)(arow + kk + ka + 4);
    v4f x2 = *(const v4f*)(arow + kk + ka + 8);
    v4f x3 = *(const v4f*)(arow + kk + ka + 12);
    v16h a;
#pragma unroll
    for (int i = 0; i < 4; ++i) {
      a[i]      = (_Float16)x0[i];
      a[i + 4]  = (_Float16)x1[i];
      a[i + 8]  = (_Float16)x2[i];
      a[i + 12] = (_Float16)x3[i];
    }
    v16h b = *(const v16h*)(bcol + kk + kb);       // 2x ds_load_b128
    c = __builtin_amdgcn_wmma_f32_16x16x32_f16(false, a, false, b, (short)0, c,
                                               false, false);
  }

  const int rbase = row0 + (hi ? 8 : 0);
  if (row0 + 16 <= N) {  // block-uniform fast path: 8 straight-line stores
#pragma unroll
    for (int r = 0; r < 8; ++r) H[(long)(rbase + r) * NC + col] = c[r];
  } else {
#pragma unroll
    for (int r = 0; r < 8; ++r) {
      int row = rbase + r;
      if (row < N) H[(long)row * NC + col] = c[r];
    }
  }
}

// Per-node attention score dots: s_src[n] = h[n,:]·a_src, s_dst[n] = h[n,:]·a_dst.
// One wave per node, shuffle reduction (wave32).
template <int F>
__global__ void node_scores(const float* __restrict__ H,
                            const float* __restrict__ asrc,
                            const float* __restrict__ adst,
                            float* __restrict__ ssrc, float* __restrict__ sdst,
                            int N) {
  int node = (int)((blockIdx.x * (long)blockDim.x + threadIdx.x) >> 5);
  int lane = threadIdx.x & 31;
  if (node >= N) return;
  float s1 = 0.f, s2 = 0.f;
#pragma unroll
  for (int f = lane; f < F; f += 32) {
    float h = H[(long)node * F + f];
    s1 += h * asrc[f];
    s2 += h * adst[f];
  }
#pragma unroll
  for (int off = 16; off > 0; off >>= 1) {
    s1 += __shfl_xor(s1, off, 32);
    s2 += __shfl_xor(s2, off, 32);
  }
  if (lane == 0) { ssrc[node] = s1; sdst[node] = s2; }
}

__device__ __forceinline__ void edge_sd(const int* __restrict__ ei, int e, int E,
                                        int& s, int& d) {
  if (e < E) { s = ei[e]; d = ei[E + e]; } else { s = d = e - E; }
}

__device__ __forceinline__ float edge_score(const float* __restrict__ ssrc,
                                            const float* __restrict__ sdst,
                                            int s, int d) {
  float sc = ssrc[s] + sdst[d];
  return sc > 0.f ? sc : NEG_SLOPE * sc;
}

__global__ void edge_max(const int* __restrict__ ei, int E, int N,
                         const float* __restrict__ ssrc,
                         const float* __restrict__ sdst,
                         unsigned* __restrict__ mcode) {
  int e = blockIdx.x * blockDim.x + threadIdx.x;
  if (e >= E + N) return;
  int s, d;
  edge_sd(ei, e, E, s, d);
  atomicMax(&mcode[d], fenc(edge_score(ssrc, sdst, s, d)));
}

__global__ void edge_denom(const int* __restrict__ ei, int E, int N,
                           const float* __restrict__ ssrc,
                           const float* __restrict__ sdst,
                           const unsigned* __restrict__ mcode,
                           float* __restrict__ denom) {
  int e = blockIdx.x * blockDim.x + threadIdx.x;
  if (e >= E + N) return;
  int s, d;
  edge_sd(ei, e, E, s, d);
  float w = __expf(edge_score(ssrc, sdst, s, d) - fdec(mcode[d]));
  atomicAdd(&denom[d], w);
}

// out[dst,:] += alpha_e * h[src,:]; one wave per edge, F/32 atomics per lane.
template <int F>
__global__ void edge_scatter(const int* __restrict__ ei, int E, int N,
                             const float* __restrict__ ssrc,
                             const float* __restrict__ sdst,
                             const unsigned* __restrict__ mcode,
                             const float* __restrict__ denom,
                             const float* __restrict__ H,
                             float* __restrict__ out) {
  int e = (int)((blockIdx.x * (long)blockDim.x + threadIdx.x) >> 5);
  int lane = threadIdx.x & 31;
  if (e >= E + N) return;
  int s, d;
  edge_sd(ei, e, E, s, d);
  float coef =
      __expf(edge_score(ssrc, sdst, s, d) - fdec(mcode[d])) / denom[d];
#pragma unroll
  for (int f = lane; f < F; f += 32)
    atomicAdd(&out[(long)d * F + f], coef * H[(long)s * F + f]);
}

template <int F, bool ELU>
__global__ void bias_act(const float* __restrict__ acc,
                         const float* __restrict__ b, float* __restrict__ out,
                         long total) {
  long i = (long)blockIdx.x * blockDim.x + threadIdx.x;
  if (i >= total) return;
  float v = acc[i] + b[i % F];
  if (ELU) v = v > 0.f ? v : expm1f(v);
  out[i] = v;
}

static inline int cdivl(long a, long b) { return (int)((a + b - 1) / b); }

extern "C" void kernel_launch(void* const* d_in, const int* in_sizes, int n_in,
                              void* d_out, int out_size, void* d_ws,
                              size_t ws_size, hipStream_t stream) {
  const float* x   = (const float*)d_in[0];
  const int*   ei  = (const int*)d_in[1];
  const float* W1  = (const float*)d_in[2];
  const float* a1s = (const float*)d_in[3];
  const float* a1d = (const float*)d_in[4];
  const float* b1  = (const float*)d_in[5];
  const float* W2  = (const float*)d_in[6];
  const float* a2s = (const float*)d_in[7];
  const float* a2d = (const float*)d_in[8];
  const float* b2  = (const float*)d_in[9];
  float* out = (float*)d_out;

  const int IN = 256, HID = 128, OUTF = 64;
  const int N = in_sizes[0] / IN;
  const int E = in_sizes[1] / 2;
  const int M = E + N;  // + self loops

  // Workspace (floats): h[N*HID] | acc[N*HID] | ssrc[N] | sdst[N] | mcode[N] | denom[N]
  float*    h     = (float*)d_ws;             // h1_pre, later ELU(out1+b1)
  float*    acc   = h + (long)N * HID;        // out1 accumulator, later h2_pre
  float*    ssrc  = acc + (long)N * HID;
  float*    sdst  = ssrc + N;
  unsigned* mcode = (unsigned*)(sdst + N);
  float*    denom = (float*)(mcode + N);

  // Dynamic LDS: A-strip f32 + W^T f16.
  const int lds1 = 16 * 256 * 4 + 128 * 256 * 2;  // 80 KB
  const int lds2 = 16 * 128 * 4 + 64 * 128 * 2;   // 24 KB
  (void)hipFuncSetAttribute((const void*)gemm_wmma<256, 128>,
                            hipFuncAttributeMaxDynamicSharedMemorySize, lds1);
  (void)hipFuncSetAttribute((const void*)gemm_wmma<128, 64>,
                            hipFuncAttributeMaxDynamicSharedMemorySize, lds2);

  dim3 b256(256);

  // ---------------- layer 1 ----------------
  zero_f32<<<cdivl((long)N * HID, 256), b256, 0, stream>>>(acc, (long)N * HID);
  zero_f32<<<cdivl(2L * N, 256), b256, 0, stream>>>((float*)mcode, 2L * N);
  gemm_wmma<256, 128><<<cdivl(N, 16), 32 * (128 / 16), lds1, stream>>>(
      x, W1, h, N);
  node_scores<128><<<cdivl((long)N * 32, 256), b256, 0, stream>>>(h, a1s, a1d,
                                                                  ssrc, sdst, N);
  edge_max<<<cdivl(M, 256), b256, 0, stream>>>(ei, E, N, ssrc, sdst, mcode);
  edge_denom<<<cdivl(M, 256), b256, 0, stream>>>(ei, E, N, ssrc, sdst, mcode,
                                                 denom);
  edge_scatter<128><<<cdivl((long)M * 32, 256), b256, 0, stream>>>(
      ei, E, N, ssrc, sdst, mcode, denom, h, acc);
  bias_act<128, true><<<cdivl((long)N * HID, 256), b256, 0, stream>>>(
      acc, b1, h, (long)N * HID);  // h := ELU(acc + b1)

  // ---------------- layer 2 ----------------
  zero_f32<<<cdivl((long)N * OUTF, 256), b256, 0, stream>>>(out, (long)N * OUTF);
  zero_f32<<<cdivl(2L * N, 256), b256, 0, stream>>>((float*)mcode, 2L * N);
  gemm_wmma<128, 64><<<cdivl(N, 16), 32 * (64 / 16), lds2, stream>>>(
      h, W2, acc, N);  // acc := h2_pre
  node_scores<64><<<cdivl((long)N * 32, 256), b256, 0, stream>>>(acc, a2s, a2d,
                                                                 ssrc, sdst, N);
  edge_max<<<cdivl(M, 256), b256, 0, stream>>>(ei, E, N, ssrc, sdst, mcode);
  edge_denom<<<cdivl(M, 256), b256, 0, stream>>>(ei, E, N, ssrc, sdst, mcode,
                                                 denom);
  edge_scatter<64><<<cdivl((long)M * 32, 256), b256, 0, stream>>>(
      ei, E, N, ssrc, sdst, mcode, denom, acc, out);
  bias_act<64, false><<<cdivl((long)N * OUTF, 256), b256, 0, stream>>>(
      out, b2, out, (long)N * OUTF);
}
